// Attention_79362405696069
// MI455X (gfx1250) — matrix operand order repaired and checked by reading, not compiled, not run on hardware
//
#include <hip/hip_runtime.h>

#define DEVINL __device__ __forceinline__

typedef __attribute__((ext_vector_type(16))) __bf16 v16bf;
typedef __attribute__((ext_vector_type(8)))  float  v8f;

// ---------- helpers ----------

DEVINL unsigned short f2bf_bits(float f) {
    unsigned u = __builtin_bit_cast(unsigned, f);
    u += 0x7FFFu + ((u >> 16) & 1u);          // round-to-nearest-even
    return (unsigned short)(u >> 16);
}
DEVINL __bf16 bf_from_bits(unsigned short s) {
    return __builtin_bit_cast(__bf16, s);
}
// 16-bit A-matrix 16x32 layout: lanes 0-15 hold K {0..7,16..23}, lanes 16-31 hold K {8..15,24..31}
DEVINL int a_kidx(int j, int half) { return (j < 8 ? j : j + 8) + (half ? 8 : 0); }
// 16-bit B-matrix 32x16 layout: lanes 0-15 hold K {0..7,16..23}, lanes 16-31 hold K {8..15,24..31}
DEVINL int b_kidx(int j, int half) { return (j < 8 ? j : j + 8) + (half ? 8 : 0); }

// ds_swizzle group-of-32 XOR patterns (offset = xor<<10 | and_mask 0x1f)
DEVINL float swz_xor(float v, int imm_dummy) { return v; } // (unused fallback)
#define SWZ_XOR(v, imm) \
    __builtin_bit_cast(float, __builtin_amdgcn_ds_swizzle(__builtin_bit_cast(int, (v)), (imm)))

DEVINL float hmax16(float v) {               // reduce across the 16-lane half-group
    v = fmaxf(v, SWZ_XOR(v, 0x201F));        // xor 8
    v = fmaxf(v, SWZ_XOR(v, 0x101F));        // xor 4
    v = fmaxf(v, SWZ_XOR(v, 0x081F));        // xor 2
    v = fmaxf(v, SWZ_XOR(v, 0x041F));        // xor 1
    return v;
}
DEVINL float hsum16(float v) {
    v += SWZ_XOR(v, 0x201F);
    v += SWZ_XOR(v, 0x101F);
    v += SWZ_XOR(v, 0x081F);
    v += SWZ_XOR(v, 0x041F);
    return v;
}

// 16B global -> LDS via the CDNA5 async engine (ASYNCcnt-tracked)
DEVINL void async_copy16(const void* gptr, void* lptr) {
    unsigned lds_off = (unsigned)(unsigned long long)lptr;   // generic LDS addr[31:0] = LDS offset
    asm volatile("global_load_async_to_lds_b128 %0, %1, off"
                 :: "v"(lds_off), "v"(gptr) : "memory");
}
DEVINL void wait_async_le(int n) {
    if (n == 0) asm volatile("s_wait_asynccnt 0x0" ::: "memory");
    else        asm volatile("s_wait_asynccnt 0x4" ::: "memory");
}

// ---------- bf16 WMMA GEMM: C[M,N] = A[M,K] * B[K,N] (+bias) ----------
// Block tile 128x64, 8 waves (4x2), each wave 32x32 = 2x2 WMMA 16x16x32 tiles.

template<bool A_BF16, bool C_BF16, bool BIAS>
__global__ __launch_bounds__(256)
void wmma_gemm(const void* __restrict__ Ap, const float* __restrict__ B,
               const float* __restrict__ bias, void* __restrict__ Cp,
               int M, int N, int K)
{
    __shared__ unsigned short As[128][32];
    __shared__ unsigned short Bs[32][64];

    const int tid  = threadIdx.x;
    const int lane = tid & 31;
    const int half = (lane >> 4) & 1;
    const int l16  = lane & 15;
    const int wid  = tid >> 5;
    const int wm   = wid & 3;
    const int wn   = wid >> 2;

    const int m0 = blockIdx.x * 128;
    const int n0 = blockIdx.y * 64;

    const float*          A32 = (const float*)Ap;
    const unsigned short* A16 = (const unsigned short*)Ap;

    v8f acc[2][2] = {};

    for (int k0 = 0; k0 < K; k0 += 32) {
        __syncthreads();
        // stage A tile 128x32
        if constexpr (A_BF16) {
            // already bf16: async global->LDS, 16B chunks (512 total, 2/thread)
            #pragma unroll
            for (int c = tid; c < 512; c += 256) {
                int row = c >> 2;
                int kc  = (c & 3) * 8;
                async_copy16(&A16[(size_t)(m0 + row) * K + (k0 + kc)], &As[row][kc]);
            }
        } else {
            // f32 source: convert in flight
            #pragma unroll
            for (int c = tid; c < 1024; c += 256) {
                int row = c >> 3;
                int kc  = (c & 7) * 4;
                size_t gi = (size_t)(m0 + row) * K + (k0 + kc);
                unsigned short* dst = &As[row][kc];
                #pragma unroll
                for (int u = 0; u < 4; ++u) dst[u] = f2bf_bits(A32[gi + u]);
            }
        }
        // stage B tile 32x64 (convert f32 -> bf16)
        #pragma unroll
        for (int c = tid; c < 512; c += 256) {
            int row = c >> 4;
            int nc  = (c & 15) * 4;
            size_t gi = (size_t)(k0 + row) * N + (n0 + nc);
            unsigned short* dst = &Bs[row][nc];
            #pragma unroll
            for (int u = 0; u < 4; ++u) dst[u] = f2bf_bits(B[gi + u]);
        }
        if (k0 + 32 < K)
            __builtin_prefetch(&B[(size_t)(k0 + 32) * N + n0], 0, 1);
        if constexpr (A_BF16) wait_async_le(0);
        __syncthreads();

        v16bf a[2], b[2];
        #pragma unroll
        for (int ms = 0; ms < 2; ++ms) {
            int row = wm * 32 + ms * 16 + l16;
            #pragma unroll
            for (int j = 0; j < 16; ++j)
                a[ms][j] = bf_from_bits(As[row][a_kidx(j, half)]);
        }
        #pragma unroll
        for (int ns = 0; ns < 2; ++ns) {
            int col = wn * 32 + ns * 16 + l16;
            #pragma unroll
            for (int j = 0; j < 16; ++j)
                b[ns][j] = bf_from_bits(Bs[b_kidx(j, half)][col]);
        }
        #pragma unroll
        for (int ms = 0; ms < 2; ++ms)
            #pragma unroll
            for (int ns = 0; ns < 2; ++ns)
                acc[ms][ns] = __builtin_amdgcn_wmma_f32_16x16x32_bf16(
                    false, a[ms], false, b[ns], (short)0, acc[ms][ns], false, false);
    }

    // epilogue: C 16x16 f32 layout -> global
    #pragma unroll
    for (int ms = 0; ms < 2; ++ms) {
        #pragma unroll
        for (int ns = 0; ns < 2; ++ns) {
            const int col = n0 + wn * 32 + ns * 16 + l16;
            #pragma unroll
            for (int r = 0; r < 8; ++r) {
                const int row = m0 + wm * 32 + ms * 16 + r + (half ? 8 : 0);
                float v = acc[ms][ns][r];
                if (BIAS) v += bias[col];
                if (C_BF16)
                    ((unsigned short*)Cp)[(size_t)row * N + col] = f2bf_bits(v);
                else
                    ((float*)Cp)[(size_t)row * N + col] = v;
            }
        }
    }
}

// ---------- flash attention with add_zero_attn ----------
// grid: (2048/64 row tiles, 4*16 batch*head), 128 threads (4 waves x 16 q rows).
// K/V tiles double-buffered through the async global->LDS engine.

DEVINL void issue_kv_tile(const unsigned short* __restrict__ qkv, int b, int h, int key0,
                          unsigned short (*Kbuf)[64], unsigned short (*Vbuf)[64], int tid)
{
    // 2 tiles x 32 rows x 64 bf16 = 2 x 256 chunks of 16B; 4 async loads / thread
    #pragma unroll
    for (int c = tid; c < 512; c += 128) {
        const int t   = c >> 8;                 // 0 = K, 1 = V
        const int cc  = c & 255;
        const int row = cc >> 3;                // 0..31
        const int off = (cc & 7) * 8;           // 0..56, 16B aligned
        const size_t gi = ((size_t)(b * 2048 + key0 + row)) * 3072
                        + 1024 * (1 + t) + h * 64 + off;
        unsigned short* dst = (t ? Vbuf : Kbuf)[row] + off;
        async_copy16(&qkv[gi], dst);
    }
}

__global__ __launch_bounds__(128)
void attn_kernel(const unsigned short* __restrict__ qkv, unsigned short* __restrict__ attn)
{
    constexpr int   NSEQ  = 2048, ROW3 = 3072, INNER = 1024;
    constexpr int   NT    = NSEQ / 32;
    constexpr float SCALE = 0.125f;                 // 64^-0.5

    __shared__ unsigned short Ks[2][32][64];
    __shared__ unsigned short Vs[2][32][64];
    __shared__ unsigned short Ps[4][16][32];

    const int tid  = threadIdx.x;
    const int lane = tid & 31;
    const int half = (lane >> 4) & 1;
    const int l16  = lane & 15;
    const int wid  = tid >> 5;

    const int bh = blockIdx.y;
    const int b  = bh >> 4;
    const int h  = bh & 15;
    const int q0 = blockIdx.x * 64 + wid * 16;      // this wave's q-row base in sequence

    // kick off tile 0 while we set up Q
    issue_kv_tile(qkv, b, h, 0, Ks[0], Vs[0], tid);

    // load Q as A fragments once (two 32-wide K chunks over d=64)
    v16bf aq[2];
    {
        const size_t qbase = ((size_t)(b * NSEQ + q0 + l16)) * ROW3 + h * 64;
        #pragma unroll
        for (int kc = 0; kc < 2; ++kc)
            #pragma unroll
            for (int j = 0; j < 16; ++j)
                aq[kc][j] = bf_from_bits(qkv[qbase + 32 * kc + a_kidx(j, half)]);
    }

    float m_run[8], l_run[8];
    #pragma unroll
    for (int r = 0; r < 8; ++r) { m_run[r] = -1e30f; l_run[r] = 0.f; }
    v8f o[4] = {};

    for (int kt = 0; kt < NT; ++kt) {
        const int cur = kt & 1;
        if (kt + 1 < NT) {
            issue_kv_tile(qkv, b, h, (kt + 1) * 32, Ks[cur ^ 1], Vs[cur ^ 1], tid);
            wait_async_le(4);                   // only current tile's 4 loads must be done
        } else {
            wait_async_le(0);
        }
        __syncthreads();                        // all waves' async data landed in LDS

        // S = Q K^T (16 q rows x 32 keys)
        v8f s[2] = {};
        #pragma unroll
        for (int kc = 0; kc < 2; ++kc) {
            #pragma unroll
            for (int ns = 0; ns < 2; ++ns) {
                v16bf bk;
                #pragma unroll
                for (int j = 0; j < 16; ++j)
                    bk[j] = bf_from_bits(Ks[cur][ns * 16 + l16][32 * kc + b_kidx(j, half)]);
                s[ns] = __builtin_amdgcn_wmma_f32_16x16x32_bf16(
                    false, aq[kc], false, bk, (short)0, s[ns], false, false);
            }
        }

        // online softmax update (row stats live replicated across the 16-lane group)
        float alpha[8];
        #pragma unroll
        for (int r = 0; r < 8; ++r) {
            const float v0 = s[0][r] * SCALE;
            const float v1 = s[1][r] * SCALE;
            const float mt = hmax16(fmaxf(v0, v1));
            const float mn = fmaxf(m_run[r], mt);
            alpha[r] = __expf(m_run[r] - mn);
            m_run[r] = mn;
            const float p0 = __expf(v0 - mn);
            const float p1 = __expf(v1 - mn);
            s[0][r] = p0; s[1][r] = p1;
            l_run[r] = l_run[r] * alpha[r] + hsum16(p0 + p1);
        }
        #pragma unroll
        for (int t = 0; t < 4; ++t)
            #pragma unroll
            for (int r = 0; r < 8; ++r)
                o[t][r] *= alpha[r];

        // re-layout P: C-layout -> LDS row-major -> A fragment (wave-private region,
        // in-wave DS ordering via dscnt is sufficient)
        #pragma unroll
        for (int ns = 0; ns < 2; ++ns)
            #pragma unroll
            for (int r = 0; r < 8; ++r)
                Ps[wid][r + (half ? 8 : 0)][ns * 16 + l16] = f2bf_bits(s[ns][r]);

        v16bf ap;
        #pragma unroll
        for (int j = 0; j < 16; ++j)
            ap[j] = bf_from_bits(Ps[wid][l16][a_kidx(j, half)]);

        #pragma unroll
        for (int t = 0; t < 4; ++t) {
            v16bf bv;
            #pragma unroll
            for (int j = 0; j < 16; ++j)
                bv[j] = bf_from_bits(Vs[cur][b_kidx(j, half)][t * 16 + l16]);
            o[t] = __builtin_amdgcn_wmma_f32_16x16x32_bf16(
                false, ap, false, bv, (short)0, o[t], false, false);
        }

        __syncthreads();                        // everyone done with buf[cur] before reuse
    }

    // add_zero_attn: extra key with logit 0 and zero value -> only denominator term
    float inv_l[8];
    #pragma unroll
    for (int r = 0; r < 8; ++r)
        inv_l[r] = 1.0f / (l_run[r] + __expf(-m_run[r]));

    #pragma unroll
    for (int t = 0; t < 4; ++t) {
        const int col = h * 64 + t * 16 + l16;
        #pragma unroll
        for (int r = 0; r < 8; ++r) {
            const int row = b * NSEQ + q0 + r + (half ? 8 : 0);
            attn[(size_t)row * INNER + col] = f2bf_bits(o[t][r] * inv_l[r]);
        }
    }
}

// ---------- launch ----------

extern "C" void kernel_launch(void* const* d_in, const int* in_sizes, int n_in,
                              void* d_out, int out_size, void* d_ws, size_t ws_size,
                              hipStream_t stream)
{
    (void)in_sizes; (void)n_in; (void)out_size; (void)ws_size;

    const float* x     = (const float*)d_in[0];   // [4,2048,1024]
    const float* W_qkv = (const float*)d_in[1];   // [1024,3072]
    const float* W_out = (const float*)d_in[2];   // [1024,1024]
    const float* b_out = (const float*)d_in[3];   // [1024]

    constexpr int Mrows = 4 * 2048;   // 8192
    constexpr int N1    = 3 * 1024;   // 3072
    constexpr int K1    = 1024;
    constexpr int N2    = 1024;

    unsigned short* qkv  = (unsigned short*)d_ws;                 // bf16 [8192,3072]
    unsigned short* attn = qkv + (size_t)Mrows * N1;              // bf16 [8192,1024]

    dim3 g1(Mrows / 128, N1 / 64);
    wmma_gemm<false, true, false><<<g1, 256, 0, stream>>>(
        x, W_qkv, nullptr, qkv, Mrows, N1, K1);

    attn_kernel<<<dim3(2048 / 64, 4 * 16), 128, 0, stream>>>(qkv, attn);

    dim3 g2(Mrows / 128, N2 / 64);
    wmma_gemm<true, false, true><<<g2, 256, 0, stream>>>(
        attn, W_out, b_out, d_out, Mrows, N2, K1);
}
